// NeuS_32899449487791
// MI455X (gfx1250) — matrix operand order repaired and checked by reading, not compiled, not run on hardware
//
#include <hip/hip_runtime.h>
#include <math.h>

// ---------------------------------------------------------------------------
// NeuS renderer for MI455X (gfx1250), wave32 + WMMA f32_16x16x32_f16.
// Forward-mode JVP: each point contributes 4 rows (value + 3 tangents) so the
// SDF gradient rides through the same WMMA GEMMs as the forward pass.
// Transcendentals use native v_exp_f32 / v_log_f32 / v_rcp_f32 / v_sin_f32.
// ---------------------------------------------------------------------------

typedef __attribute__((ext_vector_type(16))) _Float16 v16h;
typedef __attribute__((ext_vector_type(8)))  float    v8f;
typedef __attribute__((ext_vector_type(4)))  unsigned int u4;

union AB16 { v16h v; u4 q[2]; };

#define NWAVES 8
#define TM_PTS 32          // points per block
#define MROWS  128         // 32 points * (1 value + 3 tangent) rows

#define L2E 1.44269504088896340f
#define LN2 0.69314718055994531f

// ---- packed-weight offsets (in _Float16 elements) -------------------------
// tile = 32 lanes * 16 f16 = 512 elements, tiles = KT*NT
constexpr size_t O_S0 = 0;                 // 39->256 : KT2  NT16 : 16384
constexpr size_t O_S1 = 16384;             // 256->256: KT8  NT16 : 65536
constexpr size_t O_S2 = 81920;
constexpr size_t O_S3 = 147456;            // 256->217: KT8  NT14 : 57344
constexpr size_t O_S4 = 204800;
constexpr size_t O_S5 = 270336;
constexpr size_t O_S6 = 335872;
constexpr size_t O_S7 = 401408;
constexpr size_t O_S8 = 466944;            // 256->257: KT8  NT17 : 69632
constexpr size_t O_C0 = 536576;            // 301->256: KT10 NT16 : 81920
constexpr size_t O_C1 = 618496;
constexpr size_t O_C2 = 684032;
constexpr size_t O_C3 = 749568;
constexpr size_t O_C4 = 815104;            // 256->3  : KT8  NT1  : 4096
constexpr size_t F16_ELEMS = 819200;
constexpr size_t A_OFF = F16_ELEMS * 2;            // alpha: 65536 f32
constexpr size_t C_OFF = A_OFF + 65536 * 4;        // color: 65536*3 f32

struct Params {
    const float* rays_o;
    const float* rays_d;
    const float* z_vals;
    const float* variance;
    const _Float16* wpk;
    const float* sb[9];
    const float* cb[5];
    float* alpha_g;
    float* color_g;
};

// ---- native-speed transcendentals -----------------------------------------
__device__ __forceinline__ float fexp2(float x) { return __builtin_amdgcn_exp2f(x); }
__device__ __forceinline__ float flog2(float x) { return __builtin_amdgcn_logf(x); }
__device__ __forceinline__ float frcp(float x)  { return __builtin_amdgcn_rcpf(x); }
__device__ __forceinline__ float sigmoid_fast(float z) {        // 1/(1+e^-z)
    return frcp(1.0f + fexp2(-z * L2E));
}

// ---------------------------------------------------------------------------
// Weight pack kernel: W (dout,din) f32 -> per-B-fragment-lane f16 tiles.
// B order used here (32x16 KxN, contiguous k, same as A): lane l -> col n = l%16, K half = (l/16)*16,
// element e of the v16h holds K = khalf + e.
// ---------------------------------------------------------------------------
__global__ void pack_weights(const float* __restrict__ W, _Float16* __restrict__ dst,
                             int din, int dout, int KT, int NT)
{
    int idx = blockIdx.x * blockDim.x + threadIdx.x;
    int total = KT * NT * 32;
    if (idx >= total) return;
    int lane = idx & 31;
    int tile = idx >> 5;
    int nt = tile % NT;
    int kt = tile / NT;
    int n  = nt * 16 + (lane & 15);
    int kb = kt * 32 + ((lane >> 4) << 4);
    _Float16* o = dst + (size_t)idx * 16;
#pragma unroll
    for (int e = 0; e < 16; e++) {
        int k = kb + e;
        float v = (k < din && n < dout) ? W[(size_t)n * din + k] : 0.0f;
        o[e] = (_Float16)v;
    }
}

// ---------------------------------------------------------------------------
// Fused GEMM + epilogue.
//  MODE 0: softplus(100h)/100 on value rows, tangent *= sigmoid(100h_val)
//  MODE 1: final SDF layer -> sdfL / gradsL / CIN feat columns
//  MODE 2: relu (color hidden, M=32 value rows)
//  MODE 3: sigmoid, dout=3 -> global color
// ---------------------------------------------------------------------------
template<int KT, int MT, int NT, int MODE>
__device__ __forceinline__ void gemm_tile(
    const _Float16* __restrict__ X, int xstride,
    const _Float16* __restrict__ Wp,
    const float* __restrict__ bias, int dout,
    _Float16* __restrict__ Y, int ystride,
    float* sdfL, float* gradsL, _Float16* CIN, float* colOut)
{
    const int lane  = threadIdx.x & 31;
    const int w     = threadIdx.x >> 5;
    const int mt    = w % MT;
    const int mrow  = mt * 16 + (lane & 15);          // A row for this lane
    const int mbase = mt * 16 + ((lane >> 4) << 3);   // first C row for this lane

    // A fragments (contiguous k order used here, matches packed B): lane chunk [c0..c0+15], c0 = (lane/16)*16
    v16h A[KT];
#pragma unroll
    for (int kt = 0; kt < KT; kt++) {
        const _Float16* p = X + (size_t)mrow * xstride + kt * 32 + ((lane >> 4) << 4);
        AB16 u;
        u.q[0] = *(const u4*)p;
        u.q[1] = *(const u4*)(p + 8);
        A[kt] = u.v;
    }

    for (int nt = w / MT; nt < NT; nt += (NWAVES / MT)) {
        v8f acc = {0.f, 0.f, 0.f, 0.f, 0.f, 0.f, 0.f, 0.f};
#pragma unroll
        for (int kt = 0; kt < KT; kt++) {
            const _Float16* bp = Wp + ((size_t)(kt * NT + nt) * 32 + lane) * 16;
            AB16 u;
            u.q[0] = *(const u4*)bp;
            u.q[1] = *(const u4*)(bp + 8);
            acc = __builtin_amdgcn_wmma_f32_16x16x32_f16(
                false, A[kt], false, u.v, (short)0, acc, false, false);
        }
        const int   n  = nt * 16 + (lane & 15);
        const float bn = (n < dout) ? bias[n] : 0.0f;

        if (MODE == 0) {
#pragma unroll
            for (int r = 0; r < 8; r += 4) {
                float h0 = acc[r] + bn;                       // value row (c==0)
                float z  = 100.0f * h0;
                // softplus(z)/100, branchless: clamp arg, select linear branch
                float e  = fexp2(fminf(z, 20.0f) * L2E);
                float sp = flog2(1.0f + e) * (LN2 * 0.01f);
                sp = (z > 20.0f) ? h0 : sp;
                float sg = frcp(1.0f + fexp2(-z * L2E));      // softplus' = sigmoid
                Y[(size_t)(mbase + r    ) * ystride + n] = (_Float16)sp;
                Y[(size_t)(mbase + r + 1) * ystride + n] = (_Float16)(acc[r + 1] * sg);
                Y[(size_t)(mbase + r + 2) * ystride + n] = (_Float16)(acc[r + 2] * sg);
                Y[(size_t)(mbase + r + 3) * ystride + n] = (_Float16)(acc[r + 3] * sg);
            }
        } else if (MODE == 1) {
#pragma unroll
            for (int r = 0; r < 8; r++) {
                int m = mbase + r, p_ = m >> 2, c = m & 3;
                float h = acc[r] + (c == 0 ? bn : 0.0f);      // tangents get no bias
                if (n == 0) {
                    if (c == 0) sdfL[p_] = h;
                    else        gradsL[p_ * 3 + (c - 1)] = h;
                } else if (c == 0 && n <= 256) {
                    CIN[p_ * 320 + 44 + n] = (_Float16)h;     // feat -> cols 45..300
                }
            }
        } else if (MODE == 2) {
#pragma unroll
            for (int r = 0; r < 8; r++)
                Y[(size_t)(mbase + r) * ystride + n] = (_Float16)fmaxf(acc[r] + bn, 0.0f);
        } else { // MODE 3
#pragma unroll
            for (int r = 0; r < 8; r++) {
                if (n < 3) {
                    float h = acc[r] + bn;
                    colOut[(mbase + r) * 3 + n] = sigmoid_fast(h);
                }
            }
        }
    }
}

// ---------------------------------------------------------------------------
// Main kernel: one block = 32 points (128 JVP rows), 8 waves.
// ---------------------------------------------------------------------------
__global__ __launch_bounds__(256, 1) void neus_main(Params P)
{
    __shared__ _Float16 Xb0[MROWS * 256];
    __shared__ _Float16 Xb1[MROWS * 256];
    __shared__ _Float16 EMB[MROWS * 64];      // PE embedding + its Jacobian rows
    __shared__ _Float16 CIN[TM_PTS * 320];    // color-net input (K padded to 320)
    __shared__ float sdfL[TM_PTS], gradsL[TM_PTS * 3], dirsL[TM_PTS * 3], distsL[TM_PTS];

    const int tid   = threadIdx.x;
    const int pbase = blockIdx.x * TM_PTS;

    // ---- setup: points, dists, positional encoding + tangent rows ----------
    if (tid < MROWS) {
        int row = tid, p = row >> 2, c = row & 3;
        int pg = pbase + p, ray = pg >> 7, s = pg & 127;
        float ox = P.rays_o[ray * 3 + 0], oy = P.rays_o[ray * 3 + 1], oz = P.rays_o[ray * 3 + 2];
        float dx = P.rays_d[ray * 3 + 0], dy = P.rays_d[ray * 3 + 1], dz = P.rays_d[ray * 3 + 2];
        float z0 = P.z_vals[ray * 128 + s];
        float dist = (s < 127) ? (P.z_vals[ray * 128 + s + 1] - z0) : (2.0f / 128.0f);
        float mid = z0 + 0.5f * dist;
        float pt[3] = {ox + dx * mid, oy + dy * mid, oz + dz * mid};
        float e[39];
        if (c == 0) {
            e[0] = pt[0]; e[1] = pt[1]; e[2] = pt[2];
            float f = 1.0f;
            for (int m = 0; m < 6; m++) {
                for (int j = 0; j < 3; j++) {
                    e[3 + 6 * m + j]     = __sinf(pt[j] * f);
                    e[3 + 6 * m + 3 + j] = __cosf(pt[j] * f);
                }
                f *= 2.0f;
            }
        } else {                                      // d(embed)/d pt[i]
            for (int k = 0; k < 39; k++) e[k] = 0.0f;
            int i = c - 1;
            e[i] = 1.0f;
            float f = 1.0f;
            for (int m = 0; m < 6; m++) {
                float a = pt[i] * f;
                e[3 + 6 * m + i]     =  f * __cosf(a);
                e[3 + 6 * m + 3 + i] = -f * __sinf(a);
                f *= 2.0f;
            }
        }
        for (int k = 0; k < 64; k++)
            EMB[row * 64 + k] = (k < 39) ? (_Float16)e[k] : (_Float16)0.0f;
        if (c == 0) {
            dirsL[p * 3 + 0] = dx; dirsL[p * 3 + 1] = dy; dirsL[p * 3 + 2] = dz;
            distsL[p] = dist;
        }
    }
    __syncthreads();

    // ---- SDF net (layers 0..8), ping-pong Xb0 <-> Xb1 ----------------------
    const _Float16* W = P.wpk;
    gemm_tile<2, 8, 16, 0>(EMB, 64,  W + O_S0, P.sb[0], 256, Xb0, 256, sdfL, gradsL, CIN, nullptr);
    __syncthreads();
    gemm_tile<8, 8, 16, 0>(Xb0, 256, W + O_S1, P.sb[1], 256, Xb1, 256, sdfL, gradsL, CIN, nullptr);
    __syncthreads();
    gemm_tile<8, 8, 16, 0>(Xb1, 256, W + O_S2, P.sb[2], 256, Xb0, 256, sdfL, gradsL, CIN, nullptr);
    __syncthreads();
    gemm_tile<8, 8, 14, 0>(Xb0, 256, W + O_S3, P.sb[3], 217, Xb1, 256, sdfL, gradsL, CIN, nullptr);
    __syncthreads();
    // skip connection: h = concat(h[0:217], emb) / sqrt(2), applied to all JVP rows
    for (int ii = tid; ii < MROWS * 256; ii += blockDim.x) {
        int r = ii >> 8, cc = ii & 255;
        float v = (cc < 217) ? (float)Xb1[ii] : (float)EMB[r * 64 + (cc - 217)];
        Xb1[ii] = (_Float16)(v * 0.70710678f);
    }
    __syncthreads();
    gemm_tile<8, 8, 16, 0>(Xb1, 256, W + O_S4, P.sb[4], 256, Xb0, 256, sdfL, gradsL, CIN, nullptr);
    __syncthreads();
    gemm_tile<8, 8, 16, 0>(Xb0, 256, W + O_S5, P.sb[5], 256, Xb1, 256, sdfL, gradsL, CIN, nullptr);
    __syncthreads();
    gemm_tile<8, 8, 16, 0>(Xb1, 256, W + O_S6, P.sb[6], 256, Xb0, 256, sdfL, gradsL, CIN, nullptr);
    __syncthreads();
    gemm_tile<8, 8, 16, 0>(Xb0, 256, W + O_S7, P.sb[7], 256, Xb1, 256, sdfL, gradsL, CIN, nullptr);
    __syncthreads();
    gemm_tile<8, 8, 17, 1>(Xb1, 256, W + O_S8, P.sb[8], 257, Xb0, 256, sdfL, gradsL, CIN, nullptr);
    __syncthreads();

    // ---- color-net input head: [emb | dirs | normals(=grads)] + zero pad ---
    for (int ii = tid; ii < TM_PTS * 45; ii += blockDim.x) {
        int p_ = ii / 45, cc = ii % 45;
        float v;
        if (cc < 39)      v = (float)EMB[(p_ * 4) * 64 + cc];
        else if (cc < 42) v = dirsL[p_ * 3 + (cc - 39)];
        else              v = gradsL[p_ * 3 + (cc - 42)];
        CIN[p_ * 320 + cc] = (_Float16)v;
    }
    for (int ii = tid; ii < TM_PTS * 19; ii += blockDim.x) {
        int p_ = ii / 19, cc = 301 + ii % 19;
        CIN[p_ * 320 + cc] = (_Float16)0.0f;
    }
    __syncthreads();

    // ---- color net (M = 32 value rows) -------------------------------------
    float* colOut = P.color_g + (size_t)pbase * 3;
    gemm_tile<10, 2, 16, 2>(CIN, 320, W + O_C0, P.cb[0], 256, Xb0, 256, sdfL, gradsL, CIN, nullptr);
    __syncthreads();
    gemm_tile<8, 2, 16, 2>(Xb0, 256, W + O_C1, P.cb[1], 256, Xb1, 256, sdfL, gradsL, CIN, nullptr);
    __syncthreads();
    gemm_tile<8, 2, 16, 2>(Xb1, 256, W + O_C2, P.cb[2], 256, Xb0, 256, sdfL, gradsL, CIN, nullptr);
    __syncthreads();
    gemm_tile<8, 2, 16, 2>(Xb0, 256, W + O_C3, P.cb[3], 256, Xb1, 256, sdfL, gradsL, CIN, nullptr);
    __syncthreads();
    gemm_tile<8, 2, 1, 3>(Xb1, 256, W + O_C4, P.cb[4], 3, Xb0, 256, sdfL, gradsL, CIN, colOut);

    // ---- NeuS alpha per point ----------------------------------------------
    if (tid < TM_PTS) {
        int p_ = tid;
        float g0 = gradsL[p_ * 3], g1 = gradsL[p_ * 3 + 1], g2 = gradsL[p_ * 3 + 2];
        float tc = dirsL[p_ * 3] * g0 + dirsL[p_ * 3 + 1] * g1 + dirsL[p_ * 3 + 2] * g2;
        float ic = fminf(tc, 0.0f);                   // -relu(-true_cos)
        float inv_s = fexp2(P.variance[0] * 10.0f * L2E);
        float sd = sdfL[p_], dd = distsL[p_];
        float pc = sigmoid_fast((sd - ic * dd * 0.5f) * inv_s);
        float nc = sigmoid_fast((sd + ic * dd * 0.5f) * inv_s);
        float a  = (pc - nc + 1e-5f) / (pc + 1e-5f);
        P.alpha_g[pbase + p_] = fminf(fmaxf(a, 0.0f), 1.0f);
    }
}

// ---------------------------------------------------------------------------
// Per-ray transmittance cumprod + color accumulation (512 rays).
// ---------------------------------------------------------------------------
__global__ void render_rays(const float* __restrict__ alpha,
                            const float* __restrict__ col,
                            float* __restrict__ out)
{
    int ray = blockIdx.x * blockDim.x + threadIdx.x;
    if (ray >= 512) return;
    float T = 1.0f, c0 = 0.0f, c1 = 0.0f, c2 = 0.0f;
    for (int s = 0; s < 128; s++) {
        int i = ray * 128 + s;
        float a = alpha[i];
        float w = a * T;
        c0 += w * col[i * 3 + 0];
        c1 += w * col[i * 3 + 1];
        c2 += w * col[i * 3 + 2];
        T *= (1.0f - a + 1e-7f);
    }
    out[ray * 3 + 0] = c0;
    out[ray * 3 + 1] = c1;
    out[ray * 3 + 2] = c2;
}

// ---------------------------------------------------------------------------
extern "C" void kernel_launch(void* const* d_in, const int* in_sizes, int n_in,
                              void* d_out, int out_size, void* d_ws, size_t ws_size,
                              hipStream_t stream)
{
    (void)in_sizes; (void)n_in; (void)out_size; (void)ws_size;
    _Float16* wpk   = (_Float16*)d_ws;
    float* alpha_g  = (float*)((char*)d_ws + A_OFF);
    float* color_g  = (float*)((char*)d_ws + C_OFF);

    struct LW { int widx, din, dout, KT, NT; size_t off; };
    static const LW lws[14] = {
        { 3,  39, 256,  2, 16, O_S0}, { 5, 256, 256, 8, 16, O_S1},
        { 7, 256, 256,  8, 16, O_S2}, { 9, 256, 217, 8, 14, O_S3},
        {11, 256, 256,  8, 16, O_S4}, {13, 256, 256, 8, 16, O_S5},
        {15, 256, 256,  8, 16, O_S6}, {17, 256, 256, 8, 16, O_S7},
        {19, 256, 257,  8, 17, O_S8},
        {21, 301, 256, 10, 16, O_C0}, {23, 256, 256, 8, 16, O_C1},
        {25, 256, 256,  8, 16, O_C2}, {27, 256, 256, 8, 16, O_C3},
        {29, 256,   3,  8,  1, O_C4},
    };
    for (int i = 0; i < 14; i++) {
        const LW& L = lws[i];
        int total = L.KT * L.NT * 32;
        pack_weights<<<(total + 255) / 256, 256, 0, stream>>>(
            (const float*)d_in[L.widx], wpk + L.off, L.din, L.dout, L.KT, L.NT);
    }

    Params P;
    P.rays_o   = (const float*)d_in[0];
    P.rays_d   = (const float*)d_in[1];
    P.z_vals   = (const float*)d_in[2];
    for (int l = 0; l < 9; l++) P.sb[l] = (const float*)d_in[4 + 2 * l];
    for (int l = 0; l < 5; l++) P.cb[l] = (const float*)d_in[22 + 2 * l];
    P.variance = (const float*)d_in[31];
    P.wpk      = wpk;
    P.alpha_g  = alpha_g;
    P.color_g  = color_g;

    neus_main<<<65536 / TM_PTS, 256, 0, stream>>>(P);
    render_rays<<<2, 256, 0, stream>>>(alpha_g, color_g, (float*)d_out);
}
